// Graph_Discriminator_29832842838352
// MI455X (gfx1250) — matrix operand order repaired and checked by reading, not compiled, not run on hardware
//
#include <hip/hip_runtime.h>
#include <math.h>

#define B_    64
#define N_    75
#define H_    64
#define ALPHA 0.2f

typedef __attribute__((ext_vector_type(16))) _Float16 v16h;
typedef __attribute__((ext_vector_type(8)))  _Float16 v8h;
typedef __attribute__((ext_vector_type(8)))  float    v8f;

__device__ __forceinline__ float leaky(float v) { return v > 0.f ? v : ALPHA * v; }

// h[b][n][c] = (c<3) ? x[b][n][c] : 0
__global__ void init_h_kernel(const float* __restrict__ x, float* __restrict__ h) {
  int idx = blockIdx.x * blockDim.x + threadIdx.x;
  if (idx >= B_ * N_ * H_) return;
  int c = idx & 63;
  int bn = idx >> 6;
  h[idx] = (c < 3) ? x[bn * 3 + c] : 0.f;
}

// P[bn][c] = fe1_w[c][0:64] . h[bn] + fe1_b[c]   (bias folded into P)
// Q[bn][c] = fe1_w[c][64:128] . h[bn]
__global__ void __launch_bounds__(256)
pq_kernel(const float* __restrict__ h, const float* __restrict__ w1,
          const float* __restrict__ b1, float* __restrict__ P, float* __restrict__ Q) {
  __shared__ float hrow[64];
  int bn = blockIdx.x;
  int t = threadIdx.x;
  if (t < 64) hrow[t] = h[bn * 64 + t];
  __syncthreads();
  int c = t & 127;
  const float* wr = w1 + c * 129 + ((t < 128) ? 0 : 64);
  float s = (t < 128) ? b1[c] : 0.f;
#pragma unroll 16
  for (int k = 0; k < 64; ++k) s += wr[k] * hrow[k];
  if (t < 128) P[(size_t)bn * 128 + c] = s;
  else         Q[(size_t)bn * 128 + c] = s;
}

// Per (b,i): build fp16 edge-activation tile A[80][128] in LDS, then
// fe2 GEMM via v_wmma_f32_16x16x32_f16, leaky, masked sum over j.
__global__ void __launch_bounds__(256)
edge_kernel(const float* __restrict__ h, const float* __restrict__ P,
            const float* __restrict__ Q, const float* __restrict__ w1,
            const float* __restrict__ b2f, const float* __restrict__ w2,
            float* __restrict__ Asum) {
  __shared__ __align__(16) _Float16 Ash[80 * 136]; // row stride 136 halves -> padded rows for the v8h A-fragment loads
  __shared__ float Pl[128];
  __shared__ float w1d[128];
  __shared__ float distl[80];

  int blk = blockIdx.x;
  int b = blk / N_;
  int i = blk % N_;
  int t = threadIdx.x;

  if (t < 128) {
    Pl[t]  = P[((size_t)b * N_ + i) * 128 + t];
    w1d[t] = w1[t * 129 + 128];           // fe1_w[:,128] = dist column
  } else if (t < 208) {
    int j = t - 128;
    float d = 0.f;
    if (j < N_) {
      float hi0 = h[((size_t)b * N_ + i) * 64 + 0];
      float hi1 = h[((size_t)b * N_ + i) * 64 + 1];
      float hj0 = h[((size_t)b * N_ + j) * 64 + 0];
      float hj1 = h[((size_t)b * N_ + j) * 64 + 1];
      float dx = hj0 - hi0 + 1e-12f;
      float dy = hj1 - hi1 + 1e-12f;
      d = sqrtf(dx * dx + dy * dy);
    }
    distl[j] = d;
  }
  __syncthreads();

  // Fused fe1 epilogue: act(j,c) = leaky(P[c] + Q[j][c] + w1d[c]*dist[j]); rows 75..79 -> 0
  const float* Qb = Q + (size_t)b * N_ * 128;
#pragma unroll
  for (int itr = 0; itr < 40; ++itr) {
    int idx = itr * 256 + t;
    int j = idx >> 7, c = idx & 127;
    float a = 0.f;
    if (j < N_) a = leaky(Pl[c] + Qb[j * 128 + c] + w1d[c] * distl[j]);
    Ash[j * 136 + c] = (_Float16)a;
  }
  __syncthreads();

  int lane = t & 31, wv = t >> 5;     // 8 waves, wave wv owns N-tile wv
  int g = lane >> 4, ln = lane & 15;
  int col = wv * 16 + ln;             // output channel of fe2

  // B fragments: lane = N col, 16 contiguous K per lane (K = kt*32 + g*16 + e)
  v16h bfrag[4];
#pragma unroll
  for (int kt = 0; kt < 4; ++kt) {
    const float* wr = w2 + (size_t)col * 128 + kt * 32 + g * 16;
#pragma unroll
    for (int e = 0; e < 16; ++e) bfrag[kt][e] = (_Float16)wr[e];
  }
  float bias = b2f[col];
  float colsum = 0.f;

#pragma unroll
  for (int jt = 0; jt < 5; ++jt) {
    v8f acc = {};
    int m = jt * 16 + ln;             // A lane = M row
#pragma unroll
    for (int kt = 0; kt < 4; ++kt) {
      // A layout: elems 0..15 -> K = kt*32 + g*16 + e
      const v8h lo = *(const v8h*)&Ash[m * 136 + kt * 32 + g * 16];
      const v8h hi = *(const v8h*)&Ash[m * 136 + kt * 32 + g * 16 + 8];
      v16h a = __builtin_shufflevector(lo, hi, 0, 1, 2, 3, 4, 5, 6, 7, 8, 9, 10, 11, 12, 13, 14, 15);
      acc = __builtin_amdgcn_wmma_f32_16x16x32_f16(false, a, false, bfrag[kt],
                                                   (short)0, acc, false, false);
    }
    // D layout: VGPR r -> M = r + g*8; apply bias+leaky, accumulate valid rows
#pragma unroll
    for (int r = 0; r < 8; ++r) {
      int jm = jt * 16 + g * 8 + r;
      float v = leaky(acc[r] + bias);
      colsum += (jm < N_) ? v : 0.f;
    }
  }
  colsum += __shfl_xor(colsum, 16, 32); // combine the two M-halves (same column)
  if (lane < 16) Asum[((size_t)b * N_ + i) * 128 + col] = colsum;
}

// Per-node MLP: [Asum ; h] (192) -> 256 -> 256 -> tanh(64)
__global__ void __launch_bounds__(256)
node_kernel(const float* __restrict__ h, const float* __restrict__ Asum,
            const float* __restrict__ w1a, const float* __restrict__ b1a,
            const float* __restrict__ w1b, const float* __restrict__ b1b,
            const float* __restrict__ w2, const float* __restrict__ b2,
            float* __restrict__ hout) {
  __shared__ float va[256];
  __shared__ float vb[256];
  int bn = blockIdx.x;
  int t = threadIdx.x;
  if (t < 128)      va[t] = Asum[(size_t)bn * 128 + t];
  else if (t < 192) va[t] = h[(size_t)bn * 64 + (t - 128)];
  __syncthreads();
  {
    const float* wr = w1a + (size_t)t * 192;
    float s = b1a[t];
#pragma unroll 8
    for (int k = 0; k < 192; ++k) s += wr[k] * va[k];
    vb[t] = leaky(s);
  }
  __syncthreads();
  {
    const float* wr = w1b + (size_t)t * 256;
    float s = b1b[t];
#pragma unroll 8
    for (int k = 0; k < 256; ++k) s += wr[k] * vb[k];
    __syncthreads();
    va[t] = leaky(s);
  }
  __syncthreads();
  if (t < 64) {
    const float* wr = w2 + (size_t)t * 256;
    float s = b2[t];
#pragma unroll 8
    for (int k = 0; k < 256; ++k) s += wr[k] * va[k];
    hout[(size_t)bn * 64 + t] = tanhf(s);
  }
}

// out[b] = sigmoid(mean_n h[b][n][0])
__global__ void out_kernel(const float* __restrict__ h, float* __restrict__ out) {
  int b = threadIdx.x;
  if (b < B_) {
    float s = 0.f;
    for (int n = 0; n < N_; ++n) s += h[((size_t)b * N_ + n) * 64];
    s /= (float)N_;
    out[b] = 1.f / (1.f + expf(-s));
  }
}

extern "C" void kernel_launch(void* const* d_in, const int* in_sizes, int n_in,
                              void* d_out, int out_size, void* d_ws, size_t ws_size,
                              hipStream_t stream) {
  const float* x     = (const float*)d_in[0];
  const float* fe1w  = (const float*)d_in[1];  // (2,128,129)
  const float* fe1b  = (const float*)d_in[2];  // (2,128)
  const float* fe2w  = (const float*)d_in[3];  // (2,128,128)
  const float* fe2b  = (const float*)d_in[4];  // (2,128)
  const float* fn1aw = (const float*)d_in[5];  // (2,256,192)
  const float* fn1ab = (const float*)d_in[6];  // (2,256)
  const float* fn1bw = (const float*)d_in[7];  // (2,1,256,256)
  const float* fn1bb = (const float*)d_in[8];  // (2,1,256)
  const float* fn2w  = (const float*)d_in[9];  // (2,64,256)
  const float* fn2b  = (const float*)d_in[10]; // (2,64)

  float* ws = (float*)d_ws;
  float* hA = ws;                 // 64*75*64   = 307200
  float* hB = hA + 307200;        // 307200
  float* P  = hB + 307200;        // 64*75*128  = 614400
  float* Q  = P  + 614400;        // 614400
  float* As = Q  + 614400;        // 614400

  init_h_kernel<<<(B_ * N_ * H_ + 255) / 256, 256, 0, stream>>>(x, hA);

  float* hcur = hA;
  float* hnxt = hB;
  for (int it = 0; it < 2; ++it) {
    const float* w1 = fe1w + (size_t)it * 128 * 129;
    pq_kernel<<<B_ * N_, 256, 0, stream>>>(hcur, w1, fe1b + it * 128, P, Q);
    edge_kernel<<<B_ * N_, 256, 0, stream>>>(hcur, P, Q, w1, fe2b + it * 128,
                                             fe2w + (size_t)it * 128 * 128, As);
    node_kernel<<<B_ * N_, 256, 0, stream>>>(hcur, As,
                                             fn1aw + (size_t)it * 256 * 192, fn1ab + it * 256,
                                             fn1bw + (size_t)it * 256 * 256, fn1bb + it * 256,
                                             fn2w + (size_t)it * 64 * 256, fn2b + it * 64, hnxt);
    float* tmp = hcur; hcur = hnxt; hnxt = tmp;
  }
  out_kernel<<<1, 256, 0, stream>>>(hcur, (float*)d_out);
}
